// MambaBackboneComponent_11536282157350
// MI455X (gfx1250) — compile-verified, matrix operand order checked by reading, not run on hardware
//
#include <hip/hip_runtime.h>
#include <math.h>

// ---------------------------------------------------------------------------
// CDNA5 WMMA types (wave32): v_wmma_f32_16x16x32_bf16
// ---------------------------------------------------------------------------
typedef __attribute__((ext_vector_type(16))) __bf16 v16bf;
typedef __attribute__((ext_vector_type(8)))  float  v8f;

union FragAB { v16bf v; unsigned int w[8]; };
union FragC  { v8f   v; float        f[8]; };

__device__ __forceinline__ float bf2f(unsigned short h) {
  return __uint_as_float(((unsigned)h) << 16);
}
__device__ __forceinline__ unsigned short f2bf(float f) {
  unsigned u = __float_as_uint(f);                       // RNE f32 -> bf16
  return (unsigned short)((u + 0x7FFFu + ((u >> 16) & 1u)) >> 16);
}

// Async global -> LDS copy, 16 bytes per lane (ASYNCcnt tracked, in-order).
__device__ __forceinline__ void async_b128_to_lds(void* lds, const void* gsrc) {
  unsigned lds_off = (unsigned)(uintptr_t)lds;            // low 32b of LDS aperture addr
  unsigned long long ga = (unsigned long long)(uintptr_t)gsrc;
  asm volatile("global_load_async_to_lds_b128 %0, %1, off"
               :: "v"(lds_off), "v"(ga) : "memory");
}
__device__ __forceinline__ void wait_async0() {
#if __has_builtin(__builtin_amdgcn_s_wait_asynccnt)
  __builtin_amdgcn_s_wait_asynccnt(0);
#else
  asm volatile("s_wait_asynccnt 0x0" ::: "memory");
#endif
}
__device__ __forceinline__ void wait_async_le3() {       // older tile done; next tile (3 ops) may fly
#if __has_builtin(__builtin_amdgcn_s_wait_asynccnt)
  __builtin_amdgcn_s_wait_asynccnt(3);
#else
  asm volatile("s_wait_asynccnt 0x3" ::: "memory");
#endif
}

#define BM 128
#define BN 64
#define BK 32

// ---------------------------------------------------------------------------
// C[M,N] = X[M,K](bf16) * W[N,K](bf16)^T, f32 accumulate via WMMA.
// Double-buffered LDS staging with GLOBAL_LOAD_ASYNC_TO_LDS_B128: tile t+1
// streams in while tile t is consumed (ASYNCcnt in-order => wait cnt<=3).
// 16B xor swizzle keeps copies b128-granular and fragment reads bank-spread.
// Output fp32 (Cf) or bf16 (Cb). epi: 0=none 1=+bias 2=+bias,softplus.
// Requires M%128==0, N%64==0, K%32==0 (holds for every GEMM here).
// ---------------------------------------------------------------------------
__global__ __launch_bounds__(256)
void gemm_bf16(const unsigned short* __restrict__ X, int ldx,
               const unsigned short* __restrict__ W, int ldw,
               float* __restrict__ Cf, unsigned short* __restrict__ Cb, int ldc,
               int K, const float* __restrict__ bias, int epi) {
  __shared__ unsigned int As[2][BM * 16];   // 2 x 8KB
  __shared__ unsigned int Bs[2][BN * 16];   // 2 x 4KB
  const int tid  = threadIdx.x;
  const int lane = tid & 31;
  const int wid  = tid >> 5;
  const int wm   = wid >> 1;     // 0..3 : 32-row slice
  const int wn   = wid & 1;      // 0..1 : 32-col slice
  const int hi   = lane >> 4;
  const int ln   = lane & 15;
  const int bm   = blockIdx.y * BM;
  const int bn   = blockIdx.x * BN;

  // per-thread staging coordinates (chunk = 16B = 8 bf16)
  const int c0row = tid >> 2;                 // A chunks 0..255 -> rows 0..63
  const int c1row = (tid + 256) >> 2;         // A chunks 256..511 -> rows 64..127
  const int ccol  = tid & 3;
  const int c0sw  = ccol ^ (c0row & 3);
  const int c1sw  = ccol ^ (c1row & 3);

  auto stage = [&](int buf, int k0) {
    async_b128_to_lds(&As[buf][c0row * 16 + c0sw * 4],
                      X + (size_t)(bm + c0row) * ldx + k0 + ccol * 8);
    async_b128_to_lds(&As[buf][c1row * 16 + c1sw * 4],
                      X + (size_t)(bm + c1row) * ldx + k0 + ccol * 8);
    async_b128_to_lds(&Bs[buf][c0row * 16 + c0sw * 4],
                      W + (size_t)(bn + c0row) * ldw + k0 + ccol * 8);
  };

  FragC acc[2][2];
#pragma unroll
  for (int i = 0; i < 2; i++)
#pragma unroll
    for (int j = 0; j < 2; j++)
#pragma unroll
      for (int r = 0; r < 8; r++) acc[i][j].f[r] = 0.f;

  stage(0, 0);                                   // prologue: tile 0 in flight

  for (int k0 = 0; k0 < K; k0 += BK) {
    const int cur  = (k0 >> 5) & 1;
    const bool more = (k0 + BK) < K;
    if (more) {
      stage(cur ^ 1, k0 + BK);                   // stream next tile into other buffer
      if (k0 + 2 * BK < K)                       // warm L2 two tiles ahead
        __builtin_prefetch((const void*)(X + (size_t)(bm + (tid >> 1)) * ldx + k0 + 2 * BK), 0, 0);
      wait_async_le3();                          // tile `cur` landed (in-order)
    } else {
      wait_async0();
    }
    __syncthreads();

    // ---- fragments per ISA 7.12.2: K pairs {0,2,4,6,16,18,20,22}+8*hi ----
    FragAB a[2], b[2];
#pragma unroll
    for (int mi = 0; mi < 2; mi++) {
      int arow = wm * 32 + mi * 16 + ln;
#pragma unroll
      for (int p = 0; p < 8; p++) {
        int k2  = (p < 4 ? p : p + 4) + 4 * hi;  // dword (bf16-pair) index
        int k2s = (k2 & 3) | ((((k2 >> 2) ^ arow) & 3) << 2);
        a[mi].w[p] = As[cur][arow * 16 + k2s];
      }
    }
#pragma unroll
    for (int ni = 0; ni < 2; ni++) {
      int brow = wn * 32 + ni * 16 + ln;
#pragma unroll
      for (int p = 0; p < 8; p++) {
        int k2  = (p < 4 ? p : p + 4) + 4 * hi;
        int k2s = (k2 & 3) | ((((k2 >> 2) ^ brow) & 3) << 2);
        b[ni].w[p] = Bs[cur][brow * 16 + k2s];
      }
    }
#pragma unroll
    for (int mi = 0; mi < 2; mi++)
#pragma unroll
      for (int ni = 0; ni < 2; ni++)
        acc[mi][ni].v = __builtin_amdgcn_wmma_f32_16x16x32_bf16(
            false, a[mi].v, false, b[ni].v, (short)0, acc[mi][ni].v, false, false);
    __syncthreads();                             // buffer free for tile cur+2
  }

  // ---- epilogue: C/D row = base + 8*hi + r, col = base + ln ----
#pragma unroll
  for (int mi = 0; mi < 2; mi++) {
#pragma unroll
    for (int ni = 0; ni < 2; ni++) {
      int gcol = bn + wn * 32 + ni * 16 + ln;
      float bv = (epi != 0 && bias) ? bias[gcol] : 0.f;
#pragma unroll
      for (int r = 0; r < 8; r++) {
        int gm  = bm + wm * 32 + mi * 16 + hi * 8 + r;
        float v = acc[mi][ni].f[r] + bv;
        if (epi == 2) v = (v > 20.f) ? v : log1pf(__expf(v));   // softplus
        if (Cb) Cb[(size_t)gm * ldc + gcol] = f2bf(v);
        else    Cf[(size_t)gm * ldc + gcol] = v;
      }
    }
  }
}

// ---------------------------------------------------------------------------
__global__ __launch_bounds__(256)
void f32_to_bf16_kernel(const float* __restrict__ in, unsigned short* __restrict__ out, int n) {
  int i = blockIdx.x * 256 + threadIdx.x;
  if (i < n) out[i] = f2bf(in[i]);
}

// ---------------------------------------------------------------------------
// Token assembly: cls + projected obs (bf16) + zeros, + sinusoidal PE -> fp32 h
// ---------------------------------------------------------------------------
__global__ __launch_bounds__(256)
void assemble_tokens(const unsigned short* __restrict__ xe,
                     const float* __restrict__ cls,
                     float* __restrict__ h) {
  int row = blockIdx.x;          // b*2048 + t*16 + s
  int s   = row & 15;
  int l   = row & 2047;
  int bt  = row >> 4;            // b*128 + t
#pragma unroll
  for (int j = 0; j < 2; j++) {
    int d = threadIdx.x + 256 * j;
    float tok;
    if (s == 0)        tok = cls[d];
    else if (s <= 13)  tok = bf2f(xe[(size_t)(bt * 13 + (s - 1)) * 512 + d]);
    else               tok = 0.f;
    int k2 = d >> 1;
    float freq = __expf((float)(2 * k2) * (-9.210340371976184f / 512.0f));
    float ang  = (float)l * freq;
    float pe   = (d & 1) ? __cosf(ang) : __sinf(ang);
    h[(size_t)row * 512 + d] = tok + pe;
  }
}

// ---------------------------------------------------------------------------
// residual accumulate + RMSNorm; fp32 residual, bf16 and/or fp32 normed output
// ---------------------------------------------------------------------------
__global__ __launch_bounds__(256)
void rmsnorm_kernel(const float* __restrict__ x, const float* __restrict__ res_in,
                    float* __restrict__ res_out,
                    unsigned short* __restrict__ out_bf, float* __restrict__ out_f,
                    const float* __restrict__ w, int add_res) {
  __shared__ float red[256];
  int row = blockIdx.x;
  int t   = threadIdx.x;
  size_t base = (size_t)row * 512;
  float a = x[base + t];
  float b = x[base + t + 256];
  if (add_res) { a += res_in[base + t]; b += res_in[base + t + 256]; }
  if (res_out) { res_out[base + t] = a; res_out[base + t + 256] = b; }
  red[t] = a * a + b * b;
  __syncthreads();
  for (int off = 128; off > 0; off >>= 1) {
    if (t < off) red[t] += red[t + off];
    __syncthreads();
  }
  float scale = rsqrtf(red[0] * (1.0f / 512.0f) + 1e-5f);
  float o0 = a * scale * w[t];
  float o1 = b * scale * w[t + 256];
  if (out_bf) { out_bf[base + t] = f2bf(o0); out_bf[base + t + 256] = f2bf(o1); }
  if (out_f)  { out_f[base + t]  = o0;       out_f[base + t + 256]  = o1; }
}

// ---------------------------------------------------------------------------
// causal depthwise conv1d (K=4) + SiLU; xz bf16 [16384,2048] -> xc bf16
// ---------------------------------------------------------------------------
__global__ __launch_bounds__(256)
void conv_silu_kernel(const unsigned short* __restrict__ xz, const float* __restrict__ cw,
                      const float* __restrict__ cb, unsigned short* __restrict__ xc) {
  int g   = blockIdx.x * 256 + threadIdx.x;   // 16384*1024
  int d   = g & 1023;
  int row = g >> 10;
  int l   = row & 2047;
  float acc = cb[d];
#pragma unroll
  for (int j = 0; j < 4; j++) {
    int ll = l - 3 + j;
    if (ll >= 0) acc += cw[d * 4 + j] * bf2f(xz[(size_t)(row - 3 + j) * 2048 + d]);
  }
  xc[(size_t)row * 1024 + d] = f2bf(acc / (1.f + __expf(-acc)));
}

// ---------------------------------------------------------------------------
// Selective scan: 16 states x 2 channels per wave32; shfl_xor tree reduce.
// Fused gate: y = (scan + xc*D) * silu(z).  All activations bf16.
// ---------------------------------------------------------------------------
__global__ __launch_bounds__(256)
void ssm_scan_kernel(const unsigned short* __restrict__ xc, const unsigned short* __restrict__ dt,
                     const unsigned short* __restrict__ dbc, const unsigned short* __restrict__ xz,
                     const float* __restrict__ a_log, const float* __restrict__ dparam,
                     unsigned short* __restrict__ y) {
  int tid  = threadIdx.x;
  int lane = tid & 31;
  int wid  = tid >> 5;
  int hi   = lane >> 4;
  int s    = lane & 15;
  int b    = blockIdx.x >> 6;
  int d    = (blockIdx.x & 63) * 16 + wid * 2 + hi;
  float An = -__expf(a_log[d * 16 + s]);
  float Dp = dparam[d];
  float hstate = 0.f;
  for (int l = 0; l < 2048; l++) {
    size_t row = (size_t)b * 2048 + l;
    float dtv = bf2f(dt[row * 1024 + d]);
    float xv  = bf2f(xc[row * 1024 + d]);
    float Bv  = bf2f(dbc[row * 64 + 32 + s]);
    float Cv  = bf2f(dbc[row * 64 + 48 + s]);
    hstate = hstate * __expf(dtv * An) + (dtv * xv) * Bv;
    float p = hstate * Cv;
    p += __shfl_xor(p, 1, 32);
    p += __shfl_xor(p, 2, 32);
    p += __shfl_xor(p, 4, 32);
    p += __shfl_xor(p, 8, 32);
    if (s == 0) {
      float zv = bf2f(xz[row * 2048 + 1024 + d]);
      float sz = zv / (1.f + __expf(-zv));
      y[row * 1024 + d] = f2bf((p + xv * Dp) * sz);
    }
  }
}

// ---------------------------------------------------------------------------
__global__ __launch_bounds__(256)
void pool_kernel(const float* __restrict__ hf, float* __restrict__ out) {
  int m = blockIdx.x;   // 0..1023
#pragma unroll
  for (int j = 0; j < 2; j++) {
    int d = threadIdx.x + 256 * j;
    float sum = 0.f;
#pragma unroll
    for (int s = 0; s < 16; s++) sum += hf[(size_t)(m * 16 + s) * 512 + d];
    out[(size_t)m * 512 + d] = sum * (1.f / 16.f);
  }
}

// ---------------------------------------------------------------------------
extern "C" void kernel_launch(void* const* d_in, const int* in_sizes, int n_in,
                              void* d_out, int out_size, void* d_ws, size_t ws_size,
                              hipStream_t stream) {
  (void)in_sizes; (void)n_in; (void)out_size; (void)ws_size;
  const float* x    = (const float*)d_in[0];
  const float* ipw  = (const float*)d_in[1];
  const float* ipb  = (const float*)d_in[2];
  const float* cls  = (const float*)d_in[3];
  const float* nw   = (const float*)d_in[4];
  const float* inw  = (const float*)d_in[5];
  const float* cw   = (const float*)d_in[6];
  const float* cb   = (const float*)d_in[7];
  const float* xw   = (const float*)d_in[8];
  const float* dtw  = (const float*)d_in[9];
  const float* dtbp = (const float*)d_in[10];
  const float* alog = (const float*)d_in[11];
  const float* dpar = (const float*)d_in[12];
  const float* ow   = (const float*)d_in[13];
  const float* nfw  = (const float*)d_in[14];
  float* out = (float*)d_out;

  char* base = (char*)d_ws;
  size_t off = 0;
  auto alloc = [&](size_t bytes) -> void* {
    void* p = base + off;
    off = (off + bytes + 255) & ~(size_t)255;
    return p;
  };
  // bf16 weight copies (converted once per call)
  unsigned short* ipwb = (unsigned short*)alloc((size_t)512 * 128 * 2);
  unsigned short* inwb = (unsigned short*)alloc((size_t)8 * 2048 * 512 * 2);
  unsigned short* xwb  = (unsigned short*)alloc((size_t)8 * 64 * 1024 * 2);
  unsigned short* dtwb = (unsigned short*)alloc((size_t)8 * 1024 * 32 * 2);
  unsigned short* owb  = (unsigned short*)alloc((size_t)8 * 512 * 1024 * 2);
  unsigned short* xb   = (unsigned short*)alloc((size_t)13312 * 128 * 2);
  // bf16 activations
  unsigned short* xz   = (unsigned short*)alloc((size_t)16384 * 2048 * 2);
  unsigned short* xe   = xz;  // alias: xe consumed before xz first written
  unsigned short* xc   = (unsigned short*)alloc((size_t)16384 * 1024 * 2);
  unsigned short* dbc  = (unsigned short*)alloc((size_t)16384 * 64 * 2);
  unsigned short* dtact= (unsigned short*)alloc((size_t)16384 * 1024 * 2);
  unsigned short* yb   = (unsigned short*)alloc((size_t)16384 * 1024 * 2);
  unsigned short* hn   = (unsigned short*)alloc((size_t)16384 * 512 * 2);
  // fp32 residual stream
  float* hb = (float*)alloc((size_t)16384 * 512 * 4);
  float* rb = (float*)alloc((size_t)16384 * 512 * 4);
  float* hf = (float*)alloc((size_t)16384 * 512 * 4);

  dim3 blk(256);
  auto cvt = [&](const float* src, unsigned short* dst, int n) {
    f32_to_bf16_kernel<<<(n + 255) / 256, blk, 0, stream>>>(src, dst, n);
  };
  cvt(ipw, ipwb, 512 * 128);
  cvt(inw, inwb, 8 * 2048 * 512);
  cvt(xw,  xwb,  8 * 64 * 1024);
  cvt(dtw, dtwb, 8 * 1024 * 32);
  cvt(ow,  owb,  8 * 512 * 1024);
  cvt(x,   xb,   13312 * 128);

  // input projection: [13312,128] @ [512,128]^T + bias -> xe (bf16)
  gemm_bf16<<<dim3(512 / BN, 13312 / BM), blk, 0, stream>>>(
      xb, 128, ipwb, 128, nullptr, xe, 512, 128, ipb, 1);
  assemble_tokens<<<16384, blk, 0, stream>>>(xe, cls, hb);

  for (int i = 0; i < 8; i++) {
    rmsnorm_kernel<<<16384, blk, 0, stream>>>(hb, rb, rb, hn, nullptr, nw + i * 512, i > 0);
    // in_proj -> xz [16384,2048] bf16
    gemm_bf16<<<dim3(2048 / BN, 16384 / BM), blk, 0, stream>>>(
        hn, 512, inwb + (size_t)i * 2048 * 512, 512, nullptr, xz, 2048, 512, nullptr, 0);
    conv_silu_kernel<<<65536, blk, 0, stream>>>(xz, cw + (size_t)i * 1024 * 4, cb + i * 1024, xc);
    // x_proj -> dbc [16384,64] bf16
    gemm_bf16<<<dim3(64 / BN, 16384 / BM), blk, 0, stream>>>(
        xc, 1024, xwb + (size_t)i * 64 * 1024, 1024, nullptr, dbc, 64, 1024, nullptr, 0);
    // dt_proj (softplus+bias) -> dtact [16384,1024] bf16
    gemm_bf16<<<dim3(1024 / BN, 16384 / BM), blk, 0, stream>>>(
        dbc, 64, dtwb + (size_t)i * 1024 * 32, 32, nullptr, dtact, 1024, 32,
        dtbp + i * 1024, 2);
    ssm_scan_kernel<<<512, blk, 0, stream>>>(
        xc, dtact, dbc, xz, alog + (size_t)i * 1024 * 16, dpar + i * 1024, yb);
    // out_proj -> hb [16384,512] fp32 (residual stream stays fp32)
    gemm_bf16<<<dim3(512 / BN, 16384 / BM), blk, 0, stream>>>(
        yb, 1024, owb + (size_t)i * 512 * 1024, 1024, hb, nullptr, 512, 1024, nullptr, 0);
  }

  rmsnorm_kernel<<<16384, blk, 0, stream>>>(hb, rb, nullptr, nullptr, hf, nfw, 1);
  pool_kernel<<<1024, blk, 0, stream>>>(hf, out);
}
